// AttentionDecoder_54606214201413
// MI455X (gfx1250) — matrix operand order repaired and checked by reading, not compiled, not run on hardware
//
#include <hip/hip_runtime.h>
#include <hip/hip_bf16.h>
#include <math.h>

// ---------------- problem constants ----------------
constexpr int kB = 64;     // batch
constexpr int kT = 100;    // decoder steps
constexpr int kS = 400;    // encoder length
constexpr int kH = 256;    // hidden
constexpr int kE = 128;    // embedding
constexpr int kA = 612;    // attention vector size (2H + 100)
constexpr int kH2 = 2 * kH;         // 512
constexpr int kM = kB * kS;         // 25600 rows of enc_feat
constexpr int kNT = (kA + 15) / 16; // 39 tiles along A

typedef __attribute__((ext_vector_type(16))) __bf16 v16bf;
typedef __attribute__((ext_vector_type(8)))  float  v8f;
typedef __attribute__((ext_vector_type(2)))  float  v2f;

__device__ __forceinline__ float sigmoidf_(float x) {
    return 1.0f / (1.0f + __expf(-x));
}

// ---------------------------------------------------------------------------
// Kernel 0: init h,c from enc_h/enc_c; zero ctx
// ---------------------------------------------------------------------------
__global__ void init_state_kernel(const float* __restrict__ enc_h,
                                  const float* __restrict__ enc_c,
                                  float* __restrict__ h, float* __restrict__ c,
                                  float* __restrict__ ctx,
                                  float* __restrict__ hc) {
    int i = blockIdx.x * blockDim.x + threadIdx.x;
    if (i < kB * kH) { h[i] = enc_h[i]; c[i] = enc_c[i]; }
    if (i < kB * kH2) {
        ctx[i] = 0.0f;
        // hc[b, 0:H] = h, hc[b, H:2H] = c
        int b = i / kH2, j = i % kH2;
        hc[i] = (j < kH) ? enc_h[b * kH + j] : enc_c[b * kH + (j - kH)];
    }
}

// ---------------------------------------------------------------------------
// Kernel 1: enc_feat[m, n] = sum_k enc_out[m,k] * conv_W[n,k] + conv_b[n]
// M = 25600, N = 612, K = 512.  One wave32 per 16x16 C tile, bf16 WMMA.
// A-frag (16-bit A 16x32): lane l<16 -> M=l, K = 0..15;
//   lanes 16-31 -> same M rows, K = 16..31.
// B-frag (32x16): column N = lane&15; lanes 0-15 hold K=0..15,
//   lanes 16-31 hold K=16..31.
// C: acc[r] holds M = 16*tm + r + 8*(lane>>4), N = 16*tn + (lane&15).
// ---------------------------------------------------------------------------
__global__ void encfeat_wmma_kernel(const float* __restrict__ enc_out, // [M,512]
                                    const float* __restrict__ conv_W,  // [612,512]
                                    const float* __restrict__ conv_b,  // [612]
                                    float* __restrict__ enc_feat)      // [M,612]
{
    const int wave = (blockIdx.x * blockDim.x + threadIdx.x) >> 5;
    const int lane = threadIdx.x & 31;
    const int tn = wave % kNT;
    const int tm = wave / kNT;           // grid sized exactly: tm < 1600
    const int m0 = tm * 16, n0 = tn * 16;

    const int kh    = lane >> 4;
    const int mrow  = m0 + (lane & 15);
    const int ncol  = n0 + (lane & 15);
    const bool nvalid = (ncol < kA);
    const int ncol_c  = nvalid ? ncol : 0;   // clamp to avoid OOB loads

    v8f acc = {};
    const float* arow = enc_out + (size_t)mrow * kH2;
    const float* brow = conv_W + (size_t)ncol_c * kH2;

    for (int kb = 0; kb < kH2; kb += 32) {
        v16bf afrag, bfrag;
#pragma unroll
        for (int e = 0; e < 16; ++e) {
            afrag[e]     = (__bf16)arow[kb + 16 * kh + e];
        }
#pragma unroll
        for (int e = 0; e < 16; ++e) {
            float bv = brow[kb + 16 * kh + e];
            bfrag[e] = nvalid ? (__bf16)bv : (__bf16)0.0f;
        }
        acc = __builtin_amdgcn_wmma_f32_16x16x32_bf16(
            false, afrag, false, bfrag, (short)0, acc, false, false);
    }

    if (nvalid) {
        const float bias = conv_b[ncol];
        const int rbase = m0 + 8 * kh;
#pragma unroll
        for (int r = 0; r < 8; ++r)
            enc_feat[(size_t)(rbase + r) * kA + ncol] = acc[r] + bias;
    }
}

// ---------------------------------------------------------------------------
// Kernel 2 (per step): fused input x, LSTM gates, h/c update.
// One block per batch row b; 256 threads.  Writes hc = [h_new, c_new].
// ---------------------------------------------------------------------------
__global__ void step_lstm_kernel(const int*   __restrict__ ids,       // [B,T]
                                 const float* __restrict__ embedding, // [V,E]
                                 const float* __restrict__ Wx,        // [E, E+2H]
                                 const float* __restrict__ bx,        // [E]
                                 const float* __restrict__ W_ih,      // [4H,E]
                                 const float* __restrict__ b_ih,
                                 const float* __restrict__ W_hh,      // [4H,H]
                                 const float* __restrict__ b_hh,
                                 float* __restrict__ h,               // [B,H] in/out
                                 float* __restrict__ c,               // [B,H] in/out
                                 const float* __restrict__ ctx,       // [B,2H]
                                 float* __restrict__ x_out,           // [B,E]
                                 float* __restrict__ hc,              // [B,2H]
                                 int t)
{
    const int b = blockIdx.x;
    const int tid = threadIdx.x;
    __shared__ float s_in[kE + kH2];  // [emb_t, ctx] = 640
    __shared__ float s_x[kE];         // 128
    __shared__ float s_h[kH];         // 256
    __shared__ float s_c[kH];         // 256

    const int tok = ids[b * kT + t];
    for (int i = tid; i < kE; i += 256) s_in[i] = embedding[(size_t)tok * kE + i];
    for (int i = tid; i < kH2; i += 256) s_in[kE + i] = ctx[b * kH2 + i];
    for (int i = tid; i < kH; i += 256) { s_h[i] = h[b * kH + i]; s_c[i] = c[b * kH + i]; }
    __syncthreads();

    // x = [emb, ctx] @ Wx^T + bx
    if (tid < kE) {
        const float* w = Wx + (size_t)tid * (kE + kH2);
        float acc = bx[tid];
        for (int k = 0; k < kE + kH2; ++k) acc = fmaf(s_in[k], w[k], acc);
        s_x[tid] = acc;
        x_out[b * kE + tid] = acc;
    }
    __syncthreads();

    // gates (pytorch order i,f,g,o): thread tid -> indices tid + r*H
    float g4[4];
#pragma unroll
    for (int r = 0; r < 4; ++r) {
        const int j = tid + r * kH;
        const float* wih = W_ih + (size_t)j * kE;
        const float* whh = W_hh + (size_t)j * kH;
        float acc = b_ih[j] + b_hh[j];
        for (int k = 0; k < kE; ++k) acc = fmaf(s_x[k], wih[k], acc);
        for (int k = 0; k < kH; ++k) acc = fmaf(s_h[k], whh[k], acc);
        g4[r] = acc;
    }
    const float ig = sigmoidf_(g4[0]);
    const float fg = sigmoidf_(g4[1]);
    const float gg = tanhf(g4[2]);
    const float og = sigmoidf_(g4[3]);
    const float cn = fg * s_c[tid] + ig * gg;
    const float hn = og * tanhf(cn);
    h[b * kH + tid] = hn;
    c[b * kH + tid] = cn;
    hc[b * kH2 + tid] = hn;
    hc[b * kH2 + kH + tid] = cn;
}

// ---------------------------------------------------------------------------
// Kernel 3 (per step): dec_feat[b, n] = sum_k hc[b,k] * Wd[n,k] + bd[n]
// M = 64 (batch), N = 612, K = 512.  FP32 WMMA (16x16x4) keeps the serial
// recurrence in full precision; Wd slice (1.25MB) is read ONCE per step
// (weight-stationary) instead of 64x.
// FP32 A 16x4: lane l<16 -> M=l, elems K=0,1; lanes 16-31 -> K=2,3.
// FP32 B 4x16: column N = lane&15; lanes 0-15 K=0,1; lanes 16-31 K=2,3.
// One wave (32 threads) per block; 4*39 = 156 blocks.
// ---------------------------------------------------------------------------
__global__ void decfeat_wmma_kernel(const float* __restrict__ hc,  // [64,512]
                                    const float* __restrict__ Wd,  // [612,512]
                                    const float* __restrict__ bd,  // [612]
                                    float* __restrict__ dec_feat)  // [64,612]
{
    const int wave = blockIdx.x;
    const int lane = threadIdx.x & 31;
    const int tn = wave % kNT;
    const int tm = wave / kNT;           // < 4
    const int m0 = tm * 16, n0 = tn * 16;

    const int kh   = lane >> 4;
    const int mrow = m0 + (lane & 15);
    const int ncol = n0 + (lane & 15);
    const bool nvalid = (ncol < kA);
    const int ncol_c  = nvalid ? ncol : 0;

    v8f acc = {};
    const float* arow = hc + (size_t)mrow * kH2;
    const float* brow = Wd + (size_t)ncol_c * kH2;

    for (int kb = 0; kb < kH2; kb += 4) {
        v2f afrag, bfrag;
        afrag[0] = arow[kb + 2 * kh];
        afrag[1] = arow[kb + 2 * kh + 1];
        bfrag[0] = nvalid ? brow[kb + 2 * kh] : 0.0f;
        bfrag[1] = nvalid ? brow[kb + 2 * kh + 1] : 0.0f;
        acc = __builtin_amdgcn_wmma_f32_16x16x4_f32(
            false, afrag, false, bfrag, (short)0, acc, false, false);
    }

    if (nvalid) {
        const float bias = bd[ncol];
        const int rbase = m0 + 8 * kh;
#pragma unroll
        for (int r = 0; r < 8; ++r)
            dec_feat[(size_t)(rbase + r) * kA + ncol] = acc[r] + bias;
    }
}

// ---------------------------------------------------------------------------
// Kernel 4 (per step): attention scores e[b,s] = sum_a v[a]*tanh(ef + df).
// One wave32 per (b,s).  enc_feat stays L2-resident (62.7MB < 192MB L2).
// ---------------------------------------------------------------------------
__global__ void step_scores_kernel(const float* __restrict__ enc_feat, // [B*S,A]
                                   const float* __restrict__ dec_feat, // [B,A]
                                   const float* __restrict__ v,        // [A]
                                   float* __restrict__ e_out)          // [B,S]
{
    const int wave = (blockIdx.x * blockDim.x + threadIdx.x) >> 5;
    const int lane = threadIdx.x & 31;
    const int b = wave / kS;
    const int s = wave % kS;
    const float* ef = enc_feat + (size_t)(b * kS + s) * kA;
    const float* df = dec_feat + (size_t)b * kA;

    float acc = 0.0f;
    for (int a = lane; a < kA; a += 32) {
        __builtin_prefetch(ef + a + 128, 0, 0);   // global_prefetch_b8
        acc = fmaf(v[a], tanhf(ef[a] + df[a]), acc);
    }
#pragma unroll
    for (int off = 16; off > 0; off >>= 1) acc += __shfl_down(acc, off, 32);
    if (lane == 0) e_out[b * kS + s] = acc;
}

// ---------------------------------------------------------------------------
// Kernel 5 (per step): softmax over S, mask + renorm, attn store, ctx,
// p_gen, and output projection -- all per-b work fused (one block per b).
// ---------------------------------------------------------------------------
__global__ void step_attn_out_kernel(const float* __restrict__ e_in,    // [B,S]
                                     const float* __restrict__ mask,    // [B,S]
                                     const float* __restrict__ enc_out, // [B,S,2H]
                                     const float* __restrict__ h,       // [B,H]
                                     const float* __restrict__ c,       // [B,H]
                                     const float* __restrict__ x,       // [B,E]
                                     const float* __restrict__ Wpg,     // [1,1152]
                                     const float* __restrict__ bpg,     // [1]
                                     const float* __restrict__ Wout,    // [H,3H]
                                     const float* __restrict__ bout,    // [H]
                                     float* __restrict__ attn_out,      // [B,S] @ t
                                     float* __restrict__ ctx,           // [B,2H]
                                     float* __restrict__ outs,          // [B,H] @ t
                                     float* __restrict__ pgens)         // [B]   @ t
{
    const int b = blockIdx.x;
    const int tid = threadIdx.x;
    constexpr int CAT = kH2 + kH2 + kE; // 1152: [ctx, h, c, x]
    __shared__ float s_e[kS];
    __shared__ float s_cat[CAT];
    __shared__ float red[256];

    for (int s = tid; s < kS; s += 256) s_e[s] = e_in[b * kS + s];
    // preload h, c, x while softmax happens
    for (int i = tid; i < kH; i += 256) {
        s_cat[kH2 + i]      = h[b * kH + i];
        s_cat[kH2 + kH + i] = c[b * kH + i];
    }
    for (int i = tid; i < kE; i += 256) s_cat[2 * kH2 + i] = x[b * kE + i];
    __syncthreads();

    // --- softmax over S ---
    float m = -INFINITY;
    for (int s = tid; s < kS; s += 256) m = fmaxf(m, s_e[s]);
    red[tid] = m; __syncthreads();
    for (int o = 128; o > 0; o >>= 1) { if (tid < o) red[tid] = fmaxf(red[tid], red[tid + o]); __syncthreads(); }
    m = red[0]; __syncthreads();

    float sum = 0.0f;
    for (int s = tid; s < kS; s += 256) { float xv = __expf(s_e[s] - m); s_e[s] = xv; sum += xv; }
    red[tid] = sum; __syncthreads();
    for (int o = 128; o > 0; o >>= 1) { if (tid < o) red[tid] += red[tid + o]; __syncthreads(); }
    const float denom = red[0]; __syncthreads();

    // --- mask + renormalize ---
    float msum = 0.0f;
    for (int s = tid; s < kS; s += 256) {
        float xv = (s_e[s] / denom) * mask[b * kS + s];
        s_e[s] = xv; msum += xv;
    }
    red[tid] = msum; __syncthreads();
    for (int o = 128; o > 0; o >>= 1) { if (tid < o) red[tid] += red[tid + o]; __syncthreads(); }
    const float minv = 1.0f / red[0]; __syncthreads();

    for (int s = tid; s < kS; s += 256) { s_e[s] *= minv; attn_out[b * kS + s] = s_e[s]; }
    __syncthreads();

    // --- ctx[b,d] = sum_s a[s] * enc_out[b,s,d] ---
    for (int d = tid; d < kH2; d += 256) {
        const float* eo = enc_out + (size_t)b * kS * kH2 + d;
        float acc = 0.0f;
        for (int s = 0; s < kS; ++s) acc = fmaf(s_e[s], eo[(size_t)s * kH2], acc);
        s_cat[d] = acc;
        ctx[b * kH2 + d] = acc;
    }
    __syncthreads();

    // --- out[j] = bout[j] + h . Wout[j,0:H] + ctx . Wout[j,H:3H] ---
    {
        const int j = tid;
        const float* w = Wout + (size_t)j * (kH + kH2);
        float acc = bout[j];
        for (int k = 0; k < kH; ++k)  acc = fmaf(s_cat[kH2 + k], w[k], acc);
        for (int k = 0; k < kH2; ++k) acc = fmaf(s_cat[k], w[kH + k], acc);
        outs[b * kH + j] = acc;
    }

    // --- p_gen = sigmoid([ctx,h,c,x] . Wpg + bpg) ---
    float acc = 0.0f;
    for (int i = tid; i < CAT; i += 256) acc = fmaf(s_cat[i], Wpg[i], acc);
    red[tid] = acc; __syncthreads();
    for (int o = 128; o > 0; o >>= 1) { if (tid < o) red[tid] += red[tid + o]; __syncthreads(); }
    if (tid == 0) pgens[b] = sigmoidf_(red[0] + bpg[0]);
}

// ---------------------------------------------------------------------------
// Kernel 6: copy final h,c into d_out
// ---------------------------------------------------------------------------
__global__ void copy_hc_kernel(const float* __restrict__ h, const float* __restrict__ c,
                               float* __restrict__ out_h, float* __restrict__ out_c) {
    int i = blockIdx.x * blockDim.x + threadIdx.x;
    if (i < kB * kH) { out_h[i] = h[i]; out_c[i] = c[i]; }
}

// ---------------------------------------------------------------------------
extern "C" void kernel_launch(void* const* d_in, const int* in_sizes, int n_in,
                              void* d_out, int out_size, void* d_ws, size_t ws_size,
                              hipStream_t stream) {
    (void)in_sizes; (void)n_in; (void)out_size; (void)ws_size;

    // inputs in setup_inputs() order
    const int*   ids       = (const int*)  d_in[0];   // [B,T]
    const float* enc_out   = (const float*)d_in[1];   // [B,S,2H]
    const float* enc_h     = (const float*)d_in[2];   // [B,H]
    const float* enc_c     = (const float*)d_in[3];   // [B,H]
    const float* mask      = (const float*)d_in[4];   // [B,S]
    const float* embedding = (const float*)d_in[5];   // [V,E]
    const float* W_ih      = (const float*)d_in[6];
    const float* W_hh      = (const float*)d_in[7];
    const float* b_ih      = (const float*)d_in[8];
    const float* b_hh      = (const float*)d_in[9];
    const float* conv_W    = (const float*)d_in[10];  // [A,2H]
    const float* conv_b    = (const float*)d_in[11];  // [A]
    const float* v         = (const float*)d_in[12];  // [A]
    const float* Wx        = (const float*)d_in[13];  // [E,E+2H]
    const float* bx        = (const float*)d_in[14];
    const float* Wd_steps  = (const float*)d_in[15];  // [T,A,2H]
    const float* bd_steps  = (const float*)d_in[16];  // [T,A]
    const float* Wpg       = (const float*)d_in[17];  // [1,1152]
    const float* bpg       = (const float*)d_in[18];
    const float* Wout      = (const float*)d_in[19];  // [H,3H]
    const float* bout      = (const float*)d_in[20];

    // outputs (flat concat, return order): outs[T,B,H], h[B,H], c[B,H],
    // attns[T,B,S], pgens[T,B,1]
    float* out        = (float*)d_out;
    float* outs_base  = out;
    float* out_h      = outs_base + (size_t)kT * kB * kH;
    float* out_c      = out_h + (size_t)kB * kH;
    float* attns_base = out_c + (size_t)kB * kH;
    float* pgens_base = attns_base + (size_t)kT * kB * kS;

    // workspace layout (floats)
    float* ws = (float*)d_ws;
    size_t off = 0;
    float* enc_feat = ws + off; off += (size_t)kM * kA;  // 15,667,200
    float* h_st     = ws + off; off += (size_t)kB * kH;
    float* c_st     = ws + off; off += (size_t)kB * kH;
    float* ctx      = ws + off; off += (size_t)kB * kH2;
    float* hc_buf   = ws + off; off += (size_t)kB * kH2;
    float* x_buf    = ws + off; off += (size_t)kB * kE;
    float* dec_feat = ws + off; off += (size_t)kB * kA;
    float* e_buf    = ws + off; off += (size_t)kB * kS;

    // init state
    init_state_kernel<<<(kB * kH2 + 255) / 256, 256, 0, stream>>>(
        enc_h, enc_c, h_st, c_st, ctx, hc_buf);

    // one-shot enc_feat GEMM via bf16 WMMA: 1600 * 39 = 62400 wave tiles,
    // 8 waves/block -> 7800 blocks (exact, keeps EXEC all-ones for WMMA)
    encfeat_wmma_kernel<<<(kM / 16) * kNT / 8, 256, 0, stream>>>(enc_out, conv_W,
                                                                 conv_b, enc_feat);

    for (int t = 0; t < kT; ++t) {
        const float* Wd = Wd_steps + (size_t)t * kA * kH2;
        const float* bd = bd_steps + (size_t)t * kA;

        step_lstm_kernel<<<kB, 256, 0, stream>>>(
            ids, embedding, Wx, bx, W_ih, b_ih, W_hh, b_hh,
            h_st, c_st, ctx, x_buf, hc_buf, t);

        // fp32 WMMA, weight-stationary: 4 M-tiles x 39 N-tiles, 1 wave/block
        decfeat_wmma_kernel<<<4 * kNT, 32, 0, stream>>>(hc_buf, Wd, bd, dec_feat);

        // one wave per (b,s): 25600 waves = 3200 blocks of 256
        step_scores_kernel<<<kB * kS / 8, 256, 0, stream>>>(
            enc_feat, dec_feat, v, e_buf);

        step_attn_out_kernel<<<kB, 256, 0, stream>>>(
            e_buf, mask, enc_out, h_st, c_st, x_buf, Wpg, bpg, Wout, bout,
            attns_base + (size_t)t * kB * kS, ctx,
            outs_base + (size_t)t * kB * kH, pgens_base + (size_t)t * kB);
    }

    copy_hc_kernel<<<(kB * kH + 255) / 256, 256, 0, stream>>>(h_st, c_st,
                                                              out_h, out_c);
}
